// AttnBlock_35880156791647
// MI455X (gfx1250) — compile-verified, matrix operand order checked by reading, not run on hardware
//
#include <hip/hip_runtime.h>

typedef __attribute__((ext_vector_type(16))) _Float16 v16h;
typedef __attribute__((ext_vector_type(8)))  _Float16 v8h;
typedef __attribute__((ext_vector_type(8)))  float    v8f;
typedef __attribute__((ext_vector_type(4)))  int      v4i;

#define BATCH  4
#define CH     512
#define HWPX   4096
#define NGRP   32
#define CPG    16
#define HEIGHT 64.0f
#define SCALE  0.04419417382415922f  // 512^-0.5

#if __has_builtin(__builtin_amdgcn_global_load_async_to_lds_b128) && \
    __has_builtin(__builtin_amdgcn_s_wait_asynccnt)
#define USE_ASYNC_LDS 1
#else
#define USE_ASYNC_LDS 0
#endif

// ---------------------------------------------------------------------------
// 16B global -> LDS copy: async (ASYNCcnt-tracked) when available.
// Builtin signature (from clang diagnostics):
//   (v4i addrspace(1)* src, v4i addrspace(3)* dst, imm offset, imm cpol)
// ---------------------------------------------------------------------------
__device__ __forceinline__ void copy16_g2l(const _Float16* g, _Float16* l) {
#if USE_ASYNC_LDS
  __builtin_amdgcn_global_load_async_to_lds_b128(
      (__attribute__((address_space(1))) v4i*)g,
      (__attribute__((address_space(3))) v4i*)l, 0, 0);
#else
  *(v8h*)l = *(const v8h*)g;
#endif
}

template <int N>
__device__ __forceinline__ void wait_g2l() {
#if USE_ASYNC_LDS
  __builtin_amdgcn_s_wait_asynccnt(N);
#endif
}

// ---------------------------------------------------------------------------
// Fragment loader from an LDS tile with 32-half rows. CDNA5 16-bit fragment
// layout (ISA 7.12.2):
//   lanes 0-15 : row = lane,     halves 0..7 -> K 0..7,  8..15 -> K 16..23
//   lanes 16-31: row = lane-16,  halves 0..7 -> K 8..15, 8..15 -> K 24..31
// B mirrors A with "row" = output column.
// ---------------------------------------------------------------------------
__device__ __forceinline__ v16h load_frag_lds(const _Float16* p0) {
  const int lane = threadIdx.x & 31;
  const _Float16* p = p0 + (lane & 15) * 32 + ((lane & 16) ? 8 : 0);
  v8h lo = *(const v8h*)(p);
  v8h hi = *(const v8h*)(p + 16);
  v16h r;
#pragma unroll
  for (int t = 0; t < 8; ++t) { r[t] = lo[t]; r[t + 8] = hi[t]; }
  return r;
}

// Stage one K-step: A tile 64x32 halves (4KB) + B tile 128x32 halves (8KB)
// into LDS; 3 x b128 per thread (256 threads).
__device__ __forceinline__ void stage_tiles(const _Float16* __restrict__ A,
                                            int lda, int m0,
                                            const _Float16* __restrict__ Bm,
                                            int ldb, int n0, int k0,
                                            _Float16* aT, _Float16* bT) {
  const int t = threadIdx.x;
  {
    const int row = t >> 2, ch = (t & 3) * 8;  // 64 rows x 4 chunks
    copy16_g2l(A + (size_t)(m0 + row) * (size_t)lda + k0 + ch, aT + row * 32 + ch);
  }
#pragma unroll
  for (int i = 0; i < 2; ++i) {
    const int r = t + i * 256;                 // 128 rows x 4 chunks
    const int row = r >> 2, ch = (r & 3) * 8;
    copy16_g2l(Bm + (size_t)(n0 + row) * (size_t)ldb + k0 + ch, bT + row * 32 + ch);
  }
}

// ---------------------------------------------------------------------------
// WMMA GEMM: D[m,n] = sum_k A[m,k]*B[k,n].
//   A row-major (lda), batch stride abatch (0 => shared weights).
//   B stored with elem(k,n) at B[n*ldb + k], batch stride bbatch.
// Block = 256 threads = 8 waves as 2(M) x 4(N); block tile 64(M) x 128(N);
// each wave owns 32x32 = 4 accumulators. A+B double-buffered in LDS with
// async global->LDS copies overlapping the 4-WMMA compute stage.
// Epilogues:
//   0: +bias[m], f16 store transposed  T[n*CH + m]   (q_t / k_t)
//   1: +bias[m], f16 store             V[m*HWPX + n] (v, (C,HW) layout)
//   2: *scale,   f32 store             S[m*HWPX + n] (scores, row-major)
//   3:           f16 store transposed  T[n*CH + m]   (attn@V result)
//   4: +bias[m]+HEIGHT, f32 store      O[m*HWPX + n] (final output)
// ---------------------------------------------------------------------------
template <int EPI>
__global__ __launch_bounds__(256) void gemm_wmma(
    const _Float16* __restrict__ A, int lda, size_t abatch,
    const _Float16* __restrict__ Bm, int ldb, size_t bbatch,
    const float* __restrict__ bias,
    void* __restrict__ Dp, size_t dbatch,
    int K, float scale) {
  __shared__ __align__(16) _Float16 aT[2][64 * 32];
  __shared__ __align__(16) _Float16 bT[2][128 * 32];
  const int z = blockIdx.z;
  A  += (size_t)z * abatch;
  Bm += (size_t)z * bbatch;
  const int m0blk = blockIdx.y * 64;
  const int n0blk = blockIdx.x * 128;
  const int wid = threadIdx.x >> 5;
  const int wm = wid & 1;        // 0..1 -> M half
  const int wn = wid >> 1;       // 0..3 -> N quarter
  const int mbase = m0blk + wm * 32;
  const int nbase = n0blk + wn * 32;

  v8f accs[2][2] = {};

  stage_tiles(A, lda, m0blk, Bm, ldb, n0blk, 0, aT[0], bT[0]);
  int buf = 0;
  for (int k0 = 0; k0 < K; k0 += 32) {
    if (k0 + 32 < K) {
      stage_tiles(A, lda, m0blk, Bm, ldb, n0blk, k0 + 32, aT[buf ^ 1], bT[buf ^ 1]);
      wait_g2l<3>();  // async returns in-order: current tile's 3 ops landed
    } else {
      wait_g2l<0>();
    }
    __syncthreads();
    v16h a0 = load_frag_lds(aT[buf] + (wm * 32 + 0) * 32);
    v16h a1 = load_frag_lds(aT[buf] + (wm * 32 + 16) * 32);
    v16h b0 = load_frag_lds(bT[buf] + (wn * 32 + 0) * 32);
    v16h b1 = load_frag_lds(bT[buf] + (wn * 32 + 16) * 32);
    accs[0][0] = __builtin_amdgcn_wmma_f32_16x16x32_f16(false, a0, false, b0,
                                                        (short)0, accs[0][0], false, false);
    accs[0][1] = __builtin_amdgcn_wmma_f32_16x16x32_f16(false, a0, false, b1,
                                                        (short)0, accs[0][1], false, false);
    accs[1][0] = __builtin_amdgcn_wmma_f32_16x16x32_f16(false, a1, false, b0,
                                                        (short)0, accs[1][0], false, false);
    accs[1][1] = __builtin_amdgcn_wmma_f32_16x16x32_f16(false, a1, false, b1,
                                                        (short)0, accs[1][1], false, false);
    __syncthreads();
    buf ^= 1;
  }

  // D fragment layout: lane<16 -> N=lane, VGPR r -> M=r; lane>=16 -> M=8+r.
  const int lane = threadIdx.x & 31;
  const int ncol = lane & 15;
  const int moff = (lane & 16) ? 8 : 0;
#pragma unroll
  for (int mi = 0; mi < 2; ++mi) {
#pragma unroll
    for (int ni = 0; ni < 2; ++ni) {
      const v8f acc = accs[mi][ni];
      const int n  = nbase + ni * 16 + ncol;
      const int mb = mbase + mi * 16 + moff;
      if (EPI == 0 || EPI == 3) {
        _Float16* O = (_Float16*)Dp + (size_t)z * dbatch + (size_t)n * CH + mb;
        v8h h;
#pragma unroll
        for (int r = 0; r < 8; ++r) {
          float v = acc[r];
          if (EPI == 0) v += bias[mb + r];
          h[r] = (_Float16)v;
        }
        *(v8h*)O = h;  // 16B contiguous store per lane
      } else if (EPI == 1) {
        _Float16* O = (_Float16*)Dp + (size_t)z * dbatch;
#pragma unroll
        for (int r = 0; r < 8; ++r)
          O[(size_t)(mb + r) * HWPX + n] = (_Float16)(acc[r] + bias[mb + r]);
      } else if (EPI == 2) {
        float* O = (float*)Dp + (size_t)z * dbatch;
#pragma unroll
        for (int r = 0; r < 8; ++r)
          O[(size_t)(mb + r) * HWPX + n] = acc[r] * scale;
      } else {  // EPI == 4
        float* O = (float*)Dp + (size_t)z * dbatch;
#pragma unroll
        for (int r = 0; r < 8; ++r)
          O[(size_t)(mb + r) * HWPX + n] = acc[r] + bias[mb + r] + HEIGHT;
      }
    }
  }
}

// ---------------------------------------------------------------------------
// GroupNorm pass 1: per-(batch,group) mean / rstd over 16ch x 4096px
// (contiguous 65536 floats).
// ---------------------------------------------------------------------------
__global__ __launch_bounds__(256) void gn_reduce(const float* __restrict__ x,
                                                 float* __restrict__ stats) {
  const int bg = blockIdx.x;  // b*NGRP + g
  const float* base = x + (size_t)bg * (CPG * HWPX);
  float s = 0.f, s2 = 0.f;
  for (int e = threadIdx.x; e < CPG * HWPX; e += 256) {
    float v = base[e];
    s += v;
    s2 += v * v;
  }
  __shared__ float rs[256], rq[256];
  rs[threadIdx.x] = s;
  rq[threadIdx.x] = s2;
  __syncthreads();
  for (int o = 128; o > 0; o >>= 1) {
    if (threadIdx.x < o) {
      rs[threadIdx.x] += rs[threadIdx.x + o];
      rq[threadIdx.x] += rq[threadIdx.x + o];
    }
    __syncthreads();
  }
  if (threadIdx.x == 0) {
    const float invN = 1.0f / (float)(CPG * HWPX);
    float mean = rs[0] * invN;
    float var  = rq[0] * invN - mean * mean;
    stats[bg * 2]     = mean;
    stats[bg * 2 + 1] = rsqrtf(var + 1e-6f);
  }
}

// GroupNorm pass 2: normalize + affine, transpose (C,HW)->(HW,C) via LDS,
// write f16.
__global__ __launch_bounds__(256) void gn_apply(const float* __restrict__ x,
                                                const float* __restrict__ stats,
                                                const float* __restrict__ gscale,
                                                const float* __restrict__ gbias,
                                                _Float16* __restrict__ xnt) {
  __shared__ _Float16 tile[16][17];
  const int p0 = blockIdx.x * 16;
  const int g  = blockIdx.y;
  const int b  = blockIdx.z;
  const int tx = threadIdx.x & 15, ty = threadIdx.x >> 4;
  const int c = g * CPG + ty;
  const float mean = stats[(b * NGRP + g) * 2];
  const float rstd = stats[(b * NGRP + g) * 2 + 1];
  const float v = x[((size_t)(b * CH + c)) * HWPX + p0 + tx];
  tile[ty][tx] = (_Float16)((v - mean) * rstd * gscale[c] + gbias[c]);
  __syncthreads();
  xnt[((size_t)b * HWPX + p0 + ty) * CH + g * CPG + tx] = tile[tx][ty];
}

// ---------------------------------------------------------------------------
// Row softmax: one block per (b,i) row of S; values cached in 16 regs/thread;
// LDS tree reductions for max and sum; f16 output.
// ---------------------------------------------------------------------------
__global__ __launch_bounds__(256) void softmax_row(const float* __restrict__ S,
                                                   _Float16* __restrict__ attn) {
  const size_t row = blockIdx.x;
  const float* s = S + row * HWPX;
  _Float16* a = attn + row * HWPX;
  float vals[16];
  float m = -3.402823466e38f;
#pragma unroll
  for (int j = 0; j < 16; ++j) {
    float v = s[threadIdx.x + j * 256];
    vals[j] = v;
    m = fmaxf(m, v);
  }
  __shared__ float red[256];
  red[threadIdx.x] = m;
  __syncthreads();
  for (int o = 128; o > 0; o >>= 1) {
    if (threadIdx.x < o) red[threadIdx.x] = fmaxf(red[threadIdx.x], red[threadIdx.x + o]);
    __syncthreads();
  }
  m = red[0];
  __syncthreads();
  float sum = 0.f;
#pragma unroll
  for (int j = 0; j < 16; ++j) {
    float e = __expf(vals[j] - m);
    vals[j] = e;
    sum += e;
  }
  red[threadIdx.x] = sum;
  __syncthreads();
  for (int o = 128; o > 0; o >>= 1) {
    if (threadIdx.x < o) red[threadIdx.x] += red[threadIdx.x + o];
    __syncthreads();
  }
  const float inv = 1.0f / red[0];
#pragma unroll
  for (int j = 0; j < 16; ++j)
    a[threadIdx.x + j * 256] = (_Float16)(vals[j] * inv);
}

__global__ void cvt_f32_f16(const float* __restrict__ src,
                            _Float16* __restrict__ dst, int n) {
  int i = blockIdx.x * blockDim.x + threadIdx.x;
  if (i < n) dst[i] = (_Float16)src[i];
}

// ---------------------------------------------------------------------------
extern "C" void kernel_launch(void* const* d_in, const int* in_sizes, int n_in,
                              void* d_out, int out_size, void* d_ws, size_t ws_size,
                              hipStream_t stream) {
  const float* x        = (const float*)d_in[0];
  const float* gn_scale = (const float*)d_in[1];
  const float* gn_bias  = (const float*)d_in[2];
  const float* wq = (const float*)d_in[3];
  const float* bq = (const float*)d_in[4];
  const float* wk = (const float*)d_in[5];
  const float* bk = (const float*)d_in[6];
  const float* wv = (const float*)d_in[7];
  const float* bv = (const float*)d_in[8];
  const float* wp = (const float*)d_in[9];
  const float* bp = (const float*)d_in[10];
  float* out = (float*)d_out;

  char* ws = (char*)d_ws;
  size_t off = 0;
  auto carve = [&](size_t bytes) -> void* {
    void* p = ws + off;
    off = (off + bytes + 255) & ~(size_t)255;
    return p;
  };
  const size_t actN = (size_t)BATCH * HWPX * CH;  // activation elements
  _Float16* xnt  = (_Float16*)carve(actN * 2);    // (b, i, c) f16
  _Float16* qt   = (_Float16*)carve(actN * 2);    // (b, i, c) f16
  _Float16* kt   = (_Float16*)carve(actN * 2);    // (b, i, c) f16
  _Float16* vm   = (_Float16*)carve(actN * 2);    // (b, c, j) f16
  _Float16* outt = (_Float16*)carve(actN * 2);    // (b, i, c) f16
  _Float16* wqh  = (_Float16*)carve((size_t)CH * CH * 2);
  _Float16* wkh  = (_Float16*)carve((size_t)CH * CH * 2);
  _Float16* wvh  = (_Float16*)carve((size_t)CH * CH * 2);
  _Float16* wph  = (_Float16*)carve((size_t)CH * CH * 2);
  float*    stats= (float*)carve((size_t)BATCH * NGRP * 2 * 4);
  float*    Smat = (float*)carve((size_t)BATCH * HWPX * HWPX * 4);
  _Float16* attn = (_Float16*)carve((size_t)BATCH * HWPX * HWPX * 2);
  (void)ws_size; (void)in_sizes; (void)n_in; (void)out_size;

  const int wn = CH * CH;
  cvt_f32_f16<<<(wn + 255) / 256, 256, 0, stream>>>(wq, wqh, wn);
  cvt_f32_f16<<<(wn + 255) / 256, 256, 0, stream>>>(wk, wkh, wn);
  cvt_f32_f16<<<(wn + 255) / 256, 256, 0, stream>>>(wv, wvh, wn);
  cvt_f32_f16<<<(wn + 255) / 256, 256, 0, stream>>>(wp, wph, wn);

  gn_reduce<<<BATCH * NGRP, 256, 0, stream>>>(x, stats);
  gn_apply<<<dim3(HWPX / 16, NGRP, BATCH), 256, 0, stream>>>(x, stats, gn_scale,
                                                             gn_bias, xnt);

  const size_t sBatch = (size_t)HWPX * HWPX;
  const size_t aB = actN / BATCH;
  // q/k/v projections: M=CH, N=HWPX, K=CH
  gemm_wmma<0><<<dim3(HWPX / 128, CH / 64, BATCH), 256, 0, stream>>>(
      wqh, CH, 0, xnt, CH, aB, bq, qt, aB, CH, 1.0f);
  gemm_wmma<0><<<dim3(HWPX / 128, CH / 64, BATCH), 256, 0, stream>>>(
      wkh, CH, 0, xnt, CH, aB, bk, kt, aB, CH, 1.0f);
  gemm_wmma<1><<<dim3(HWPX / 128, CH / 64, BATCH), 256, 0, stream>>>(
      wvh, CH, 0, xnt, CH, aB, bv, vm, aB, CH, 1.0f);

  // scores: S[i,j] = scale * sum_c q[i,c] k[j,c] ; M=N=HWPX, K=CH
  gemm_wmma<2><<<dim3(HWPX / 128, HWPX / 64, BATCH), 256, 0, stream>>>(
      qt, CH, aB, kt, CH, aB, nullptr, Smat, sBatch, CH, SCALE);

  softmax_row<<<BATCH * HWPX, 256, 0, stream>>>(Smat, attn);

  // attn @ V: out[c,i] = sum_j v[c,j] attn[i,j] ; M=CH, N=HWPX, K=HWPX
  gemm_wmma<3><<<dim3(HWPX / 128, CH / 64, BATCH), 256, 0, stream>>>(
      vm, HWPX, aB, attn, HWPX, sBatch, nullptr, outt, aB, HWPX, 1.0f);

  // projection + height-scalar bug: O[o,i] = sum_c wp[o,c] out[c,i] + bp[o] + 64
  gemm_wmma<4><<<dim3(HWPX / 128, CH / 64, BATCH), 256, 0, stream>>>(
      wph, CH, 0, outt, CH, aB, bp, out, (size_t)CH * HWPX, CH, 1.0f);
}
